// THAN_45664092291649
// MI455X (gfx1250) — matrix operand order repaired and checked by reading, not compiled, not run on hardware
//
#include <hip/hip_runtime.h>

typedef unsigned short u16;
typedef __attribute__((ext_vector_type(16))) __bf16 v16bf;
typedef __attribute__((ext_vector_type(8)))  float  v8f;

#define TK 40           // neighbors per node
#define NB 128          // batch
#define N2TOT 256
#define N1TOT 10240     // level-0 node count (N2*K)
#define R1TOT 409600    // level-0 k-rows (N1*K)
#define CH_NODES 1024   // chunk: nodes per level-0 chunk
#define CH_ROWS  40960  // chunk: k-rows per chunk
#define NCHUNK   10

// ---------- bf16 helpers (round-to-nearest-even) ----------
__device__ __forceinline__ u16 f2b(float f) {
  union { float f; unsigned u; } v; v.f = f;
  unsigned r = v.u + 0x7FFFu + ((v.u >> 16) & 1u);
  return (u16)(r >> 16);
}
__device__ __forceinline__ float b2f(u16 h) {
  union { unsigned u; float f; } v; v.u = ((unsigned)h) << 16;
  return v.f;
}

// ---------- WMMA B-fragment swizzle for a row-major [K,N] matrix ----------
// 32x16 (KxN) tile -> 32 lanes x 16 contiguous bf16:
//   lane = (n%16) + 16*((k%32)/16), elem i = k%16
// Tiles laid out (kt-major, nt-minor) so the 4 n-tiles a block needs for one
// k-step are one contiguous 4KB span.
__device__ __forceinline__ size_t bswz(int k, int n, int N) {
  int kt = k >> 5, kk = k & 31;
  int kh = kk >> 4, i = kk & 15;
  int nt = n >> 4, nn = n & 15;
  int lane = nn + (kh << 4);
  return ((size_t)(kt * (N >> 4) + nt) << 9) + ((size_t)lane << 4) + i;
}

__device__ __forceinline__ unsigned lds_off(const void* p) {
  return (unsigned)(unsigned long long)p;
}
// one 16B lane-chunk, global -> LDS, tracked by ASYNCcnt
#define ASYNC_LDS_CP16(lo, gp)                                   \
  asm volatile("global_load_async_to_lds_b128 %0, %1, off"       \
               :: "v"(lo), "v"((unsigned long long)(gp)) : "memory")

// =====================================================================
// Generic bf16 WMMA GEMM: C[M,N] = A[M,K] * Bswz[K,N] (+bias, relu)
// BM=128, BN=64, BK=32; 256 threads = 8 waves (4x2), wave tile 32x32.
// Double-buffered LDS; all staging via async global->LDS (ASYNCcnt),
// next k-tile is in flight behind the current tile's WMMAs.
// M % 128 == 0, N % 64 == 0, K % 32 == 0 at every call site.
// =====================================================================
template<bool OF32, bool BIAS, bool RELU>
__global__ __launch_bounds__(256) void gemm_bf16_wmma(
    const u16* __restrict__ A, const u16* __restrict__ Bsw,
    const float* __restrict__ bias, void* __restrict__ Cout,
    int Mrows, int Kdim, int Ncols, int ldc)
{
  __shared__ alignas(16) u16 As[2][128 * 40];  // 128x32, stride 40 (pad)
  __shared__ alignas(16) u16 Bs[2][2048];      // swizzled 32x64 k-tile

  const int tid  = threadIdx.x;
  const int wave = tid >> 5;
  const int lane = tid & 31;
  const int wm   = wave >> 1;        // 0..3
  const int wn   = wave & 1;         // 0..1
  const int m0   = blockIdx.y * 128;
  const int n0   = blockIdx.x * 64;
  const int lm   = lane & 15;
  const int lh   = lane >> 4;
  const int ntiles = Ncols >> 4;
  const int nt0    = n0 >> 4;

  v8f acc[2][2];
  for (int i = 0; i < 2; ++i)
    for (int j = 0; j < 2; ++j)
      for (int q = 0; q < 8; ++q) acc[i][j][q] = 0.f;

  // per-thread staging coordinates (2 A chunks + 1 B chunk, 16B each)
  const int r0 = tid >> 2;            // A rows 0..63 (+64 for chunk 2)
  const int cc = (tid & 3) * 8;       // A cols 0,8,16,24
  const u16* gA0 = A + (size_t)(m0 + r0) * Kdim + cc;
  const u16* gA1 = A + (size_t)(m0 + r0 + 64) * Kdim + cc;
  const u16* gB  = Bsw + (size_t)nt0 * 512 + (size_t)tid * 8;
  const size_t bstride = (size_t)ntiles * 512;
  const unsigned la0[2] = { lds_off(&As[0][r0 * 40 + cc]),
                            lds_off(&As[1][r0 * 40 + cc]) };
  const unsigned la1[2] = { lds_off(&As[0][(r0 + 64) * 40 + cc]),
                            lds_off(&As[1][(r0 + 64) * 40 + cc]) };
  const unsigned lb[2]  = { lds_off(&Bs[0][tid * 8]),
                            lds_off(&Bs[1][tid * 8]) };

  const int nk = Kdim >> 5;
  // preload k-tile 0 into buffer 0
  ASYNC_LDS_CP16(la0[0], gA0);
  ASYNC_LDS_CP16(la1[0], gA1);
  ASYNC_LDS_CP16(lb[0],  gB);
  gA0 += 32; gA1 += 32; gB += bstride;

  for (int t = 0; t < nk; ++t) {
    const int cur = t & 1;
    if (t + 1 < nk) {
      const int nxt = cur ^ 1;
      ASYNC_LDS_CP16(la0[nxt], gA0);
      ASYNC_LDS_CP16(la1[nxt], gA1);
      ASYNC_LDS_CP16(lb[nxt],  gB);
      gA0 += 32; gA1 += 32; gB += bstride;
      // 3 ops of tile t+1 outstanding; wait for tile t's 3 only
      asm volatile("s_wait_asynccnt 0x3" ::: "memory");
    } else {
      asm volatile("s_wait_asynccnt 0x0" ::: "memory");
    }
    __syncthreads();

    // --- fragments: all contiguous 16B LDS loads ---
    v16bf afrag[2], bfrag[2];
    for (int fi = 0; fi < 2; ++fi) {
      const u16* p = &As[cur][(wm * 32 + fi * 16 + lm) * 40];
      for (int i = 0; i < 8; ++i)
        afrag[fi][i] = __builtin_bit_cast(__bf16, p[lh * 16 + i]);
      for (int i = 0; i < 8; ++i)
        afrag[fi][8 + i] = __builtin_bit_cast(__bf16, p[lh * 16 + 8 + i]);
    }
    for (int fj = 0; fj < 2; ++fj) {
      const u16* bp = &Bs[cur][((wn << 1) + fj) * 512 + lane * 16];
      for (int i = 0; i < 16; ++i)
        bfrag[fj][i] = __builtin_bit_cast(__bf16, bp[i]);
    }
    for (int fi = 0; fi < 2; ++fi)
      for (int fj = 0; fj < 2; ++fj)
        acc[fi][fj] = __builtin_amdgcn_wmma_f32_16x16x32_bf16(
            false, afrag[fi], false, bfrag[fj], (short)0, acc[fi][fj],
            false, false);
    __syncthreads();  // hand-off: nobody re-stages a buffer still being read
  }

  // epilogue: C element j of lane -> m = lh*8+j, n = lm within 16x16 tile
  for (int fi = 0; fi < 2; ++fi) {
    int rbase = m0 + wm * 32 + fi * 16 + lh * 8;
    for (int fj = 0; fj < 2; ++fj) {
      int col = n0 + wn * 32 + fj * 16 + lm;
      for (int j = 0; j < 8; ++j) {
        float v = acc[fi][fj][j];
        if (BIAS) v += bias[col];
        if (RELU) v = v > 0.f ? v : 0.f;
        size_t idx = (size_t)(rbase + j) * ldc + col;
        if (OF32) ((float*)Cout)[idx] = v;
        else      ((u16*)Cout)[idx]  = f2b(v);
      }
    }
  }
}

// =====================================================================
// Weight prep: fused KV weights  Wfused[lv][512,768] (B-swizzled out)
//   rows 0..255  : Wrel[lv][t] @ [Wk_top | Wv_top]   (t = r>>7)
//   rows 256..511: [Wk | Wv] rows 128..383
// =====================================================================
__global__ __launch_bounds__(128) void prep_kv_kernel(
    const float* __restrict__ Wk, const float* __restrict__ Wv,
    const float* __restrict__ Wrel, u16* __restrict__ WkvF)
{
  int c  = blockIdx.x * 128 + threadIdx.x;  // 0..767
  int r  = blockIdx.y;                      // 0..511
  int lv = blockIdx.z;                      // 0..1
  const float* Wsrc = (c < 384) ? (Wk + (size_t)lv * 384 * 384)
                                : (Wv + (size_t)lv * 384 * 384);
  int cc = (c < 384) ? c : c - 384;
  float v;
  if (r < 256) {
    int t = r >> 7, rr = r & 127;
    const float* wr = Wrel + (((size_t)lv * 2 + t) * 128 + rr) * 128;
    float acc = 0.f;
    for (int j = 0; j < 128; ++j) acc += wr[j] * Wsrc[(size_t)j * 384 + cc];
    v = acc;
  } else {
    v = Wsrc[(size_t)(r - 128) * 384 + cc];
  }
  WkvF[(size_t)lv * 512 * 768 + bswz(r, c, 768)] = f2b(v);
}

// convert f32 row-major [K,N] -> bf16 B-swizzled
__global__ void conv_swizzle_kernel(const float* __restrict__ s,
                                    u16* __restrict__ d, int K, int N)
{
  int i = blockIdx.x * 256 + threadIdx.x;
  if (i < K * N) {
    int k = i / N, n = i % N;
    d[bswz(k, n, N)] = f2b(s[i]);
  }
}

// =====================================================================
// Level-0 K-row builder (chunked): Xk[lrow, 512] =
//   [h0*(et==0) | h0*(et==1) | e_feat | cos(dt*freq+phase)]
// =====================================================================
__global__ __launch_bounds__(128) void build_k0_kernel(
    int rowbase,
    const int* __restrict__ nghN, const int* __restrict__ nghE,
    const int* __restrict__ nghET, const float* __restrict__ nghT,
    const float* __restrict__ t1flat,
    const float* __restrict__ n_feat, const float* __restrict__ mem_tab,
    const float* __restrict__ e_feat,
    const float* __restrict__ freq, const float* __restrict__ phase,
    u16* __restrict__ Xk)
{
  int lrow = blockIdx.x;
  int row  = rowbase + lrow;
  int d    = threadIdx.x;
  int node = nghN[row];
  int eidx = nghE[row];
  int et   = nghET[row];
  float h  = n_feat[(size_t)node * 128 + d] + mem_tab[(size_t)node * 128 + d];
  float e  = e_feat[(size_t)eidx * 128 + d];
  int   n  = row / TK;
  float dt = t1flat[n] - nghT[row];
  float te = cosf(dt * freq[d] + phase[d]);
  u16* o = Xk + (size_t)lrow * 512;
  o[d]       = (et == 0) ? f2b(h) : (u16)0;
  o[128 + d] = (et == 1) ? f2b(h) : (u16)0;
  o[256 + d] = f2b(e);
  o[384 + d] = f2b(te);
}

// Level-1 K-row builder (ngh_feat = h1 rows align 1:1 with k-rows)
__global__ __launch_bounds__(128) void build_k1_kernel(
    const u16* __restrict__ h1,
    const int* __restrict__ nghE, const int* __restrict__ nghET,
    const float* __restrict__ nghT, const float* __restrict__ cut_time,
    const float* __restrict__ e_feat,
    const float* __restrict__ freq, const float* __restrict__ phase,
    u16* __restrict__ Xk)
{
  int row = blockIdx.x;
  int d   = threadIdx.x;
  int eidx = nghE[row];
  int et   = nghET[row];
  float hv = b2f(h1[(size_t)row * 128 + d]);
  float e  = e_feat[(size_t)eidx * 128 + d];
  int n2   = row / TK;
  float dt = cut_time[n2 & 127] - nghT[row];
  float te = cosf(dt * freq[d] + phase[d]);
  u16* o = Xk + (size_t)row * 512;
  o[d]       = (et == 0) ? f2b(hv) : (u16)0;
  o[128 + d] = (et == 1) ? f2b(hv) : (u16)0;
  o[256 + d] = f2b(e);
  o[384 + d] = f2b(te);
}

// q rows: [src_feat | zeros | cos(phase)]
__global__ __launch_bounds__(128) void build_q0_kernel(
    const int* __restrict__ nghN2flat,
    const float* __restrict__ n_feat, const float* __restrict__ mem_tab,
    const float* __restrict__ phase, u16* __restrict__ q)
{
  int n = blockIdx.x, d = threadIdx.x;
  int node = nghN2flat[n];
  float s = n_feat[(size_t)node * 128 + d] + mem_tab[(size_t)node * 128 + d];
  u16* o = q + (size_t)n * 384;
  o[d]       = f2b(s);
  o[128 + d] = 0;
  o[256 + d] = f2b(cosf(phase[d]));
}

__global__ __launch_bounds__(128) void build_q1_kernel(
    const int* __restrict__ src_idx, const int* __restrict__ tgt_idx,
    const float* __restrict__ n_feat, const float* __restrict__ mem_tab,
    const float* __restrict__ phase, u16* __restrict__ q)
{
  int n = blockIdx.x, d = threadIdx.x;
  int node = (n < NB) ? src_idx[n] : tgt_idx[n - NB];
  float s = n_feat[(size_t)node * 128 + d] + mem_tab[(size_t)node * 128 + d];
  u16* o = q + (size_t)n * 384;
  o[d]       = f2b(s);
  o[128 + d] = 0;
  o[256 + d] = f2b(cosf(phase[d]));
}

// =====================================================================
// Attention per node: scores (4 heads x 40), masked softmax, weighted V
// kv rows local to chunk: [n*40+k, 768] = [kh(384) | vh(384)]
// =====================================================================
__global__ __launch_bounds__(384) void attn_kernel(
    const float* __restrict__ qh,    // [nodes, 384] (chunk-offset applied)
    const u16* __restrict__ kv,      // [nodes*40, 768] (chunk-local)
    const int* __restrict__ nidx,    // [nodes*40] neighbor node ids (offset)
    u16* __restrict__ O)             // [nodes, 384] (chunk-offset applied)
{
  __shared__ float s[4][TK];
  __shared__ float aw[4][TK];
  int n = blockIdx.x, t = threadIdx.x;
  const float scale = 0.1020620726f;  // 1/sqrt(96)

  if (t < 4 * TK) {
    int h = t / TK, k = t % TK;
    const u16*   kr = kv + ((size_t)n * TK + k) * 768 + h * 96;
    const float* qr = qh + (size_t)n * 384 + h * 96;
    float acc = 0.f;
    for (int j = 0; j < 96; ++j) acc += qr[j] * b2f(kr[j]);
    acc *= scale;
    if (nidx[n * TK + k] == 0) acc = -1e9f;
    s[h][k] = acc;
  }
  __syncthreads();
  if (t < 4) {
    float m = -3.4e38f;
    for (int k = 0; k < TK; ++k) m = fmaxf(m, s[t][k]);
    float sum = 0.f;
    for (int k = 0; k < TK; ++k) { float e = expf(s[t][k] - m); aw[t][k] = e; sum += e; }
    float inv = 1.f / sum;
    for (int k = 0; k < TK; ++k) aw[t][k] *= inv;
  }
  __syncthreads();
  {
    int h = t / 96, d = t % 96;
    const u16* vr = kv + (size_t)n * TK * 768 + 384 + h * 96 + d;
    float o = 0.f;
    for (int k = 0; k < TK; ++k) o += aw[h][k] * b2f(vr[(size_t)k * 768]);
    O[(size_t)n * 384 + t] = f2b(o);
  }
}

// copy src_feat (q cols 0..127) into MLP input cols 384..511
__global__ __launch_bounds__(128) void fill_xm_kernel(
    const u16* __restrict__ q, u16* __restrict__ Xm)
{
  int r = blockIdx.x, d = threadIdx.x;
  Xm[(size_t)r * 512 + 384 + d] = q[(size_t)r * 384 + d];
}

// final: sigmoid(se . Wmatch[et] . te + bmatch[et])
__global__ __launch_bounds__(128) void score_kernel(
    const float* __restrict__ h2, const float* __restrict__ Wmatch,
    const float* __restrict__ bmatch, const int* __restrict__ etype,
    float* __restrict__ out)
{
  __shared__ float red[128];
  int b = blockIdx.x, e = threadIdx.x;
  int et = etype[b];
  const float* se = h2 + (size_t)b * 128;
  const float* te = h2 + (size_t)(b + NB) * 128;
  const float* W  = Wmatch + (size_t)et * 128 * 128;
  float acc = 0.f;
  for (int d = 0; d < 128; ++d) acc += se[d] * W[d * 128 + e];
  red[e] = acc * te[e];
  __syncthreads();
  for (int st = 64; st > 0; st >>= 1) {
    if (e < st) red[e] += red[e + st];
    __syncthreads();
  }
  if (e == 0) {
    float x = red[0] + bmatch[et];
    out[b] = 1.f / (1.f + expf(-x));
  }
}

// =====================================================================
extern "C" void kernel_launch(void* const* d_in, const int* in_sizes, int n_in,
                              void* d_out, int out_size, void* d_ws, size_t ws_size,
                              hipStream_t stream) {
  const int*   src_idx  = (const int*)  d_in[0];
  const int*   tgt_idx  = (const int*)  d_in[1];
  const float* cut_time = (const float*)d_in[2];
  const int*   etype_l  = (const int*)  d_in[5];
  const int*   nghN2    = (const int*)  d_in[6];
  const int*   nghE2    = (const int*)  d_in[7];
  const float* nghT2    = (const float*)d_in[8];
  const int*   nghET2   = (const int*)  d_in[9];
  const int*   nghN1    = (const int*)  d_in[11];
  const int*   nghE1    = (const int*)  d_in[12];
  const float* nghT1    = (const float*)d_in[13];
  const int*   nghET1   = (const int*)  d_in[14];
  const float* n_feat   = (const float*)d_in[16];
  const float* e_feat   = (const float*)d_in[17];
  const float* mem_tab  = (const float*)d_in[18];
  const float* freq     = (const float*)d_in[19];
  const float* phase    = (const float*)d_in[20];
  const float* Wq       = (const float*)d_in[21];
  const float* Wk       = (const float*)d_in[22];
  const float* Wv       = (const float*)d_in[23];
  const float* Wo       = (const float*)d_in[24];
  const float* Wm1      = (const float*)d_in[25];
  const float* bm1      = (const float*)d_in[26];
  const float* Wm2      = (const float*)d_in[27];
  const float* bm2      = (const float*)d_in[28];
  const float* Wrel     = (const float*)d_in[29];
  const float* Wmatch   = (const float*)d_in[30];
  const float* bmatch   = (const float*)d_in[31];
  float* out = (float*)d_out;

  // ---- workspace carve ----
  char* w = (char*)d_ws;
  auto alloc = [&](size_t bytes) -> void* {
    void* p = (void*)w;
    w += (bytes + 255) & ~(size_t)255;
    return p;
  };
  u16*   q0   = (u16*)  alloc((size_t)N1TOT * 384 * 2);
  float* qh0  = (float*)alloc((size_t)N1TOT * 384 * 4);
  u16*   XkC  = (u16*)  alloc((size_t)CH_ROWS * 512 * 2);  // reused by level1
  u16*   kvC  = (u16*)  alloc((size_t)CH_ROWS * 768 * 2);  // reused by level1
  u16*   O0   = (u16*)  alloc((size_t)N1TOT * 384 * 2);
  u16*   Xm0  = (u16*)  alloc((size_t)N1TOT * 512 * 2);
  u16*   Hh0  = (u16*)  alloc((size_t)N1TOT * 128 * 2);
  u16*   h1   = (u16*)  alloc((size_t)N1TOT * 128 * 2);
  u16*   q1   = (u16*)  alloc((size_t)N2TOT * 384 * 2);
  float* qh1  = (float*)alloc((size_t)N2TOT * 384 * 4);
  u16*   O1   = (u16*)  alloc((size_t)N2TOT * 384 * 2);
  u16*   Xm1  = (u16*)  alloc((size_t)N2TOT * 512 * 2);
  u16*   Hh1  = (u16*)  alloc((size_t)N2TOT * 128 * 2);
  float* h2   = (float*)alloc((size_t)N2TOT * 128 * 4);
  u16*   WkvF = (u16*)  alloc((size_t)2 * 512 * 768 * 2);
  u16*   WqB  = (u16*)  alloc((size_t)2 * 384 * 384 * 2);
  u16*   WoB  = (u16*)  alloc((size_t)2 * 384 * 384 * 2);
  u16*   Wm1B = (u16*)  alloc((size_t)2 * 512 * 128 * 2);
  u16*   Wm2B = (u16*)  alloc((size_t)2 * 128 * 128 * 2);

  // ---- weight prep (all B operands stored pre-swizzled) ----
  prep_kv_kernel<<<dim3(6, 512, 2), 128, 0, stream>>>(Wk, Wv, Wrel, WkvF);
  for (int lv = 0; lv < 2; ++lv) {
    conv_swizzle_kernel<<<(384 * 384 + 255) / 256, 256, 0, stream>>>(
        Wq + (size_t)lv * 384 * 384, WqB + (size_t)lv * 384 * 384, 384, 384);
    conv_swizzle_kernel<<<(384 * 384 + 255) / 256, 256, 0, stream>>>(
        Wo + (size_t)lv * 384 * 384, WoB + (size_t)lv * 384 * 384, 384, 384);
    conv_swizzle_kernel<<<(512 * 128 + 255) / 256, 256, 0, stream>>>(
        Wm1 + (size_t)lv * 512 * 128, Wm1B + (size_t)lv * 512 * 128, 512, 128);
    conv_swizzle_kernel<<<(128 * 128 + 255) / 256, 256, 0, stream>>>(
        Wm2 + (size_t)lv * 128 * 128, Wm2B + (size_t)lv * 128 * 128, 128, 128);
  }

  // ---- level 0 ----
  build_q0_kernel<<<N1TOT, 128, 0, stream>>>(nghN2, n_feat, mem_tab, phase, q0);
  gemm_bf16_wmma<true, false, false><<<dim3(384 / 64, N1TOT / 128), 256, 0, stream>>>(
      q0, WqB, nullptr, qh0, N1TOT, 384, 384, 384);

  for (int c = 0; c < NCHUNK; ++c) {
    int rowbase  = c * CH_ROWS;
    int nodebase = c * CH_NODES;
    build_k0_kernel<<<CH_ROWS, 128, 0, stream>>>(
        rowbase, nghN1, nghE1, nghET1, nghT1, nghT2,
        n_feat, mem_tab, e_feat, freq, phase, XkC);
    gemm_bf16_wmma<false, false, false><<<dim3(768 / 64, CH_ROWS / 128), 256, 0, stream>>>(
        XkC, WkvF, nullptr, kvC, CH_ROWS, 512, 768, 768);
    attn_kernel<<<CH_NODES, 384, 0, stream>>>(
        qh0 + (size_t)nodebase * 384, kvC, nghN1 + (size_t)rowbase,
        O0 + (size_t)nodebase * 384);
  }

  gemm_bf16_wmma<false, false, false><<<dim3(384 / 64, N1TOT / 128), 256, 0, stream>>>(
      O0, WoB, nullptr, Xm0, N1TOT, 384, 384, 512);
  fill_xm_kernel<<<N1TOT, 128, 0, stream>>>(q0, Xm0);
  gemm_bf16_wmma<false, true, true><<<dim3(128 / 64, N1TOT / 128), 256, 0, stream>>>(
      Xm0, Wm1B, bm1, Hh0, N1TOT, 512, 128, 128);
  gemm_bf16_wmma<false, true, false><<<dim3(128 / 64, N1TOT / 128), 256, 0, stream>>>(
      Hh0, Wm2B, bm2, h1, N1TOT, 128, 128, 128);

  // ---- level 1 ----
  build_q1_kernel<<<N2TOT, 128, 0, stream>>>(src_idx, tgt_idx, n_feat, mem_tab, phase, q1);
  gemm_bf16_wmma<true, false, false><<<dim3(384 / 64, N2TOT / 128), 256, 0, stream>>>(
      q1, WqB + (size_t)384 * 384, nullptr, qh1, N2TOT, 384, 384, 384);
  build_k1_kernel<<<N1TOT, 128, 0, stream>>>(
      h1, nghE2, nghET2, nghT2, cut_time, e_feat, freq, phase, XkC);
  gemm_bf16_wmma<false, false, false><<<dim3(768 / 64, N1TOT / 128), 256, 0, stream>>>(
      XkC, WkvF + (size_t)512 * 768, nullptr, kvC, N1TOT, 512, 768, 768);
  attn_kernel<<<N2TOT, 384, 0, stream>>>(qh1, kvC, nghN2, O1);
  gemm_bf16_wmma<false, false, false><<<dim3(384 / 64, N2TOT / 128), 256, 0, stream>>>(
      O1, WoB + (size_t)384 * 384, nullptr, Xm1, N2TOT, 384, 384, 512);
  fill_xm_kernel<<<N2TOT, 128, 0, stream>>>(q1, Xm1);
  gemm_bf16_wmma<false, true, true><<<dim3(128 / 64, N2TOT / 128), 256, 0, stream>>>(
      Xm1, Wm1B + (size_t)512 * 128, bm1 + 128, Hh1, N2TOT, 512, 128, 128);
  gemm_bf16_wmma<true, true, false><<<dim3(128 / 64, N2TOT / 128), 256, 0, stream>>>(
      Hh1, Wm2B + (size_t)128 * 128, bm2 + 128, h2, N2TOT, 128, 128, 128);

  // ---- final score ----
  score_kernel<<<NB, 128, 0, stream>>>(h2, Wmatch, bmatch, etype_l, out);
}
